// AttentionLayer_33243046871392
// MI455X (gfx1250) — matrix operand order repaired and checked by reading, not compiled, not run on hardware
//
#include <hip/hip_runtime.h>
#include <hip/hip_bf16.h>
#include <math.h>

#define B_ 256
#define T_ 1024
#define F_ 512
#define U_ 256

typedef _Float16 v16h __attribute__((ext_vector_type(16)));
typedef _Float16 v8h  __attribute__((ext_vector_type(8)));
typedef _Float16 v4h  __attribute__((ext_vector_type(4)));
typedef float    v8f  __attribute__((ext_vector_type(8)));

// ---------------------------------------------------------------------------
// Kernel 1: W_o (F x U, f32, row-major) -> W_oT (U x F, f16) tiled transpose.
// Makes WMMA B-fragments contiguous 32B loads (k-major per output column).
// ---------------------------------------------------------------------------
__global__ void transpose_wo(const float* __restrict__ Wo, _Float16* __restrict__ WoT) {
    __shared__ _Float16 tile[32][33];
    const int fb = blockIdx.x * 32;
    const int ub = blockIdx.y * 32;
    const int tx = threadIdx.x & 31;
    const int ty = threadIdx.x >> 5;          // 0..7 (256 threads)
#pragma unroll
    for (int i = 0; i < 32; i += 8) {
        // read coalesced over u
        tile[ty + i][tx] = (_Float16)Wo[(size_t)(fb + ty + i) * U_ + (ub + tx)];
    }
    __syncthreads();
#pragma unroll
    for (int i = 0; i < 32; i += 8) {
        // write coalesced over f
        WoT[(size_t)(ub + ty + i) * F_ + (fb + tx)] = tile[tx][ty + i];
    }
}

// ---------------------------------------------------------------------------
// Kernel 2: sh[b,u] = h_final[b,:] . W_h[:,u] + bias[u]   (B x U, tiny GEMV)
// ---------------------------------------------------------------------------
__global__ void scoreh_kernel(const float* __restrict__ inputs,
                              const float* __restrict__ Wh,
                              const float* __restrict__ bias,
                              float* __restrict__ sh) {
    __shared__ float hrow[F_];
    const int b   = blockIdx.x;
    const int tid = threadIdx.x;              // 256 threads; tid == u
    const float* hf = inputs + ((size_t)b * T_ + (T_ - 1)) * F_;
    for (int i = tid; i < F_; i += 256) hrow[i] = hf[i];
    __syncthreads();
    float acc = bias[tid];
#pragma unroll 8
    for (int f = 0; f < F_; ++f) acc += hrow[f] * Wh[(size_t)f * U_ + tid];
    sh[b * U_ + tid] = acc;
}

// ---------------------------------------------------------------------------
// Kernel 3: main WMMA GEMM + tanh + v-dot epilogue -> logits[b,t]
//   Block: 512 threads (16 waves), owns 64 rows of the (B*T, F) A matrix
//   (halves W_oT L2 traffic vs 32-row tiles: each block reads W_oT once).
//   Wave w owns U-columns [16w, 16w+16); 16 K-tiles (K=32) x 4 row tiles
//   = 64 v_wmma_f32_16x16x32_f16 per wave.
// ---------------------------------------------------------------------------
__global__ __launch_bounds__(512)
void logits_kernel(const float* __restrict__ inputs,
                   const _Float16* __restrict__ WoT,
                   const float* __restrict__ sh,
                   const float* __restrict__ vvec,
                   float* __restrict__ logits) {
    __shared__ _Float16 Alds[64 * 512];       // 64 KB, f16 A tile
    __shared__ float    lsum[64];             // per-row logit partials

    const int tid  = threadIdx.x;
    const int row0 = blockIdx.x * 64;         // 64 | T, block never straddles b
    const int batch = row0 >> 10;             // T_ = 1024

    if (tid < 64) lsum[tid] = 0.0f;

    // ---- stage A: 64 rows x 512 cols, f32 -> f16 (fully coalesced) ----
    const float* src = inputs + (size_t)row0 * F_;
#pragma unroll
    for (int i = 0; i < 16; ++i) {
        const int e = i * 2048 + tid * 4;     // flat (row*512 + col)
        float4 x = *(const float4*)(src + e);
        v4h h = { (_Float16)x.x, (_Float16)x.y, (_Float16)x.z, (_Float16)x.w };
        *(v4h*)(&Alds[e]) = h;
    }
    __syncthreads();

    const int wave = tid >> 5;
    const int lane = tid & 31;
    const int nl   = lane & 15;               // col within N-tile / row within M-tile
    const int khi  = lane >> 4;               // half-wave K split
    const int n0   = wave * 16;               // this wave's U columns

    v8f acc[4];
#pragma unroll
    for (int rt = 0; rt < 4; ++rt) acc[rt] = (v8f){};

    // B fragment: b[j] = WoT[n0+nl][kt*32 + khi*16 + j]  (32B contiguous)
    const _Float16* bbase = WoT + (size_t)(n0 + nl) * F_ + khi * 16;
    // A fragment: a[j] = A[m][kt*32 + khi*16 + j]
    const _Float16* abase = Alds + nl * 512 + khi * 16;

#pragma unroll
    for (int kt = 0; kt < 16; ++kt) {
        v16h bf = *(const v16h*)(bbase + kt * 32);
#pragma unroll
        for (int rt = 0; rt < 4; ++rt) {
            const _Float16* ap = abase + kt * 32 + rt * (16 * 512);
            v8h alo = *(const v8h*)(ap);
            v8h ahi = *(const v8h*)(ap + 8);
            v16h a = __builtin_shufflevector(alo, ahi,
                        0,1,2,3,4,5,6,7,8,9,10,11,12,13,14,15);
            acc[rt] = __builtin_amdgcn_wmma_f32_16x16x32_f16(
                          false, a, false, bf, (short)0, acc[rt], false, false);
        }
    }

    // ---- epilogue: s = tanh(score_o + sh), logit partial = s . v ----
    // C layout: element g of acc, lane l -> row (g + 8*(l>>4)), col (l&15).
    // Reduce over the 16 columns of this wave with a shfl_xor butterfly
    // (masks 1/2/4/8 stay inside the 16-lane khi group), then one
    // conflict-free LDS atomic per row for the cross-wave combine.
    const int   u   = n0 + nl;
    const float shv = sh[batch * U_ + u];
    const float vv  = vvec[u];
#pragma unroll
    for (int rt = 0; rt < 4; ++rt) {
#pragma unroll
        for (int g = 0; g < 8; ++g) {
            float p = tanhf(acc[rt][g] + shv) * vv;
            p += __shfl_xor(p, 1, 32);
            p += __shfl_xor(p, 2, 32);
            p += __shfl_xor(p, 4, 32);
            p += __shfl_xor(p, 8, 32);
            if (nl == 0) atomicAdd(&lsum[rt * 16 + g + khi * 8], p);
        }
    }
    __syncthreads();
    if (tid < 64) logits[row0 + tid] = lsum[tid];
}

// ---------------------------------------------------------------------------
// Kernel 4: softmax over T per batch -> attention weights
// ---------------------------------------------------------------------------
__global__ void softmax_kernel(const float* __restrict__ logits,
                               float* __restrict__ aw) {
    __shared__ float red[256];
    const int b   = blockIdx.x;
    const int tid = threadIdx.x;              // 256 threads, 4 t's each
    const float* lrow = logits + (size_t)b * T_;

    float l[4];
    float mx = -3.0e38f;
#pragma unroll
    for (int i = 0; i < 4; ++i) { l[i] = lrow[i * 256 + tid]; mx = fmaxf(mx, l[i]); }
    red[tid] = mx;
    __syncthreads();
    for (int s = 128; s > 0; s >>= 1) {
        if (tid < s) red[tid] = fmaxf(red[tid], red[tid + s]);
        __syncthreads();
    }
    mx = red[0];
    __syncthreads();

    float e[4], ssum = 0.0f;
#pragma unroll
    for (int i = 0; i < 4; ++i) { e[i] = expf(l[i] - mx); ssum += e[i]; }
    red[tid] = ssum;
    __syncthreads();
    for (int s = 128; s > 0; s >>= 1) {
        if (tid < s) red[tid] += red[tid + s];
        __syncthreads();
    }
    const float inv = 1.0f / red[0];
#pragma unroll
    for (int i = 0; i < 4; ++i) aw[(size_t)b * T_ + i * 256 + tid] = e[i] * inv;
}

// ---------------------------------------------------------------------------
// Kernel 5: context[b,f] = sum_t inputs[b,t,f] * w[b,t]  (streaming, HBM-bound)
// ---------------------------------------------------------------------------
__global__ void context_kernel(const float* __restrict__ inputs,
                               const float* __restrict__ aw,
                               float* __restrict__ ctx) {
    __shared__ float wl[T_];
    const int b   = blockIdx.x;
    const int tid = threadIdx.x;              // 256 threads x float2 = 512 = F
    for (int i = tid; i < T_; i += 256) wl[i] = aw[(size_t)b * T_ + i];
    __syncthreads();
    const float* base = inputs + (size_t)b * T_ * F_ + tid * 2;
    float ax = 0.0f, ay = 0.0f;
#pragma unroll 4
    for (int t = 0; t < T_; ++t) {
        float2 x = *(const float2*)(base + (size_t)t * F_);
        ax += x.x * wl[t];
        ay += x.y * wl[t];
    }
    ctx[(size_t)b * F_ + tid * 2]     = ax;
    ctx[(size_t)b * F_ + tid * 2 + 1] = ay;
}

// ---------------------------------------------------------------------------
extern "C" void kernel_launch(void* const* d_in, const int* in_sizes, int n_in,
                              void* d_out, int out_size, void* d_ws, size_t ws_size,
                              hipStream_t stream) {
    const float* inputs = (const float*)d_in[0];   // B*T*F
    const float* Wo     = (const float*)d_in[1];   // F*U
    const float* Wh     = (const float*)d_in[2];   // F*U
    const float* bias   = (const float*)d_in[3];   // U
    const float* vvec   = (const float*)d_in[4];   // U

    float* ctx = (float*)d_out;                    // B*F
    float* aw  = (float*)d_out + B_ * F_;          // B*T

    // workspace layout (<= 1.5 MB)
    char* ws = (char*)d_ws;
    _Float16* WoT    = (_Float16*)ws;                         // F*U*2   = 256 KB
    float*    sh     = (float*)(ws + (size_t)F_ * U_ * 2);    // B*U*4   = 256 KB
    float*    logits = (float*)(ws + (size_t)F_ * U_ * 2
                                   + (size_t)B_ * U_ * 4);    // B*T*4   = 1 MB

    transpose_wo <<<dim3(F_ / 32, U_ / 32), 256, 0, stream>>>(Wo, WoT);
    scoreh_kernel<<<B_, 256, 0, stream>>>(inputs, Wh, bias, sh);
    logits_kernel<<<(B_ * T_) / 64, 512, 0, stream>>>(inputs, WoT, sh, vvec, logits);
    softmax_kernel<<<B_, 256, 0, stream>>>(logits, aw);
    context_kernel<<<B_, 256, 0, stream>>>(inputs, aw, ctx);
}
